// MixedSparseAttention_5463198401138
// MI455X (gfx1250) — compile-verified, matrix operand order checked by reading, not run on hardware
//
#include <hip/hip_runtime.h>
#include <math.h>

namespace {

constexpr int kB  = 2;
constexpr int kH  = 16;
constexpr int kQ  = 2048;
constexpr int kD  = 64;
constexpr int KB  = 32;     // KV block rows per iteration
constexpr int WAVES = 4;    // waves per workgroup
constexpr int QT  = 16;     // q rows per wave

typedef __attribute__((ext_vector_type(16))) _Float16 v16h;
typedef __attribute__((ext_vector_type(8)))  _Float16 v8h;
typedef __attribute__((ext_vector_type(8)))  float    v8f;
typedef __attribute__((ext_vector_type(4)))  float    f4;

union VFrag { v16h v; v8h h[2]; };

__device__ __forceinline__ v8f wmma_f16(v16h a, v16h b, v8f c) {
  // D(f32 16x16) = A(f16 16x32) * B(f16 32x16) + C
  return __builtin_amdgcn_wmma_f32_16x16x32_f16(
      /*neg_a=*/false, a, /*neg_b=*/false, b,
      /*c_mod=*/(short)0, c, /*reuse_a=*/false, /*reuse_b=*/false);
}

} // namespace

__global__ __launch_bounds__(WAVES * 32)
void msa_flash_fwd_kernel(const float* __restrict__ Q, const float* __restrict__ K,
                          const float* __restrict__ V, const float* __restrict__ M,
                          float* __restrict__ O) {
  const int tid  = threadIdx.x;
  const int lane = tid & 31;
  const int wave = tid >> 5;
  const int lm   = lane & 15;   // row index inside 16-wide tile
  const int hi   = lane >> 4;   // half-lane selector (0/1)

  const int bh = blockIdx.y;            // b*H + h
  const int b  = bh / kH;
  const size_t head_off = (size_t)bh * kQ * kD;
  const float* Qb = Q + head_off;
  const float* Kb = K + head_off;
  const float* Vb = V + head_off;
  float*       Ob = O + head_off;
  const float* Mb = M + (size_t)b * kQ * kQ;   // mask (B,1,Q,Q)

  const int q0   = blockIdx.x * (WAVES * QT) + wave * QT;
  const int qrow = q0 + lm;

  // LDS staging: K block row-major [k][d], V block transposed [d][k].
  // +8 half padding keeps 16B alignment and spreads LDS banks.
  __shared__ _Float16 sK [KB][kD + 8];
  __shared__ _Float16 sVT[kD][KB + 8];

  // ---- Q^T B-fragments (lane = q column, halves = d), kept all kernel ----
  VFrag qb[2];
  {
    const float* qp = Qb + (size_t)qrow * kD;
    #pragma unroll
    for (int g = 0; g < 2; ++g) {             // d groups [0,32) and [32,64)
      float qf[16];
      #pragma unroll
      for (int j = 0; j < 2; ++j) {           // half groups {hi*8..}, {16+hi*8..}
        f4 a = *(const f4*)(qp + g * 32 + j * 16 + hi * 8);
        f4 c = *(const f4*)(qp + g * 32 + j * 16 + hi * 8 + 4);
        #pragma unroll
        for (int i = 0; i < 4; ++i) { qf[j * 8 + i] = a[i]; qf[j * 8 + 4 + i] = c[i]; }
      }
      #pragma unroll
      for (int i = 0; i < 16; ++i) qb[g].v[i] = (_Float16)qf[i];
    }
  }

  float run_m = -INFINITY;
  float run_l = 0.0f;
  v8f oc[4];                                   // O^T accumulators, 4 d-tiles
  #pragma unroll
  for (int t = 0; t < 4; ++t)
    #pragma unroll
    for (int r = 0; r < 8; ++r) oc[t][r] = 0.0f;

  for (int kb = 0; kb < kQ; kb += KB) {
    // ---- cooperative stage: global f32 -> LDS f16 (coalesced b128 loads) ----
    {
      const int kr = tid >> 2;                 // 0..31  (KV row in block)
      const int dq = (tid & 3) * 16;           // 0,16,32,48
      const float* kp = Kb + (size_t)(kb + kr) * kD + dq;
      const float* vp = Vb + (size_t)(kb + kr) * kD + dq;
      float kf[16], vf[16];
      #pragma unroll
      for (int i = 0; i < 4; ++i) {
        f4 a = *(const f4*)(kp + i * 4);
        f4 c = *(const f4*)(vp + i * 4);
        #pragma unroll
        for (int j = 0; j < 4; ++j) { kf[i * 4 + j] = a[j]; vf[i * 4 + j] = c[j]; }
      }
      #pragma unroll
      for (int i = 0; i < 16; ++i) sK[kr][dq + i] = (_Float16)kf[i];
      #pragma unroll
      for (int i = 0; i < 16; ++i) sVT[dq + i][kr] = (_Float16)vf[i];   // transpose
    }
    __syncthreads();

    // ---- K A-fragments: two 16-row k subtiles x two 32-wide d halves ----
    VFrag ka[2][2];
    #pragma unroll
    for (int s = 0; s < 2; ++s) {
      const _Float16* row = &sK[s * 16 + lm][0];
      ka[s][0].h[0] = *(const v8h*)(row +      hi * 8);
      ka[s][0].h[1] = *(const v8h*)(row + 16 + hi * 8);
      ka[s][1].h[0] = *(const v8h*)(row + 32 + hi * 8);
      ka[s][1].h[1] = *(const v8h*)(row + 48 + hi * 8);
    }
    // ---- V^T A-fragments: 4 d-tiles, K-dim = 32 (whole block) ----
    VFrag va[4];
    #pragma unroll
    for (int t = 0; t < 4; ++t) {
      const _Float16* row = &sVT[t * 16 + lm][0];
      va[t].h[0] = *(const v8h*)(row +      hi * 8);
      va[t].h[1] = *(const v8h*)(row + 16 + hi * 8);
    }

    // ---- S^T = K . Q^T : two 16x16 f32 tiles (lane owns q row q0+lm) ----
    v8f c0, c1;
    #pragma unroll
    for (int r = 0; r < 8; ++r) { c0[r] = 0.0f; c1[r] = 0.0f; }
    c0 = wmma_f16(ka[0][0].v, qb[0].v, c0);
    c0 = wmma_f16(ka[0][1].v, qb[1].v, c0);
    c1 = wmma_f16(ka[1][0].v, qb[0].v, c1);
    c1 = wmma_f16(ka[1][1].v, qb[1].v, c1);

    // ---- add mask; per-lane scores: k = kb + {sub}*16 + hi*8 + r ----
    float s[16];
    {
      const float* mrow = Mb + (size_t)qrow * kQ + kb + hi * 8;
      f4 m0 = *(const f4*)(mrow);
      f4 m1 = *(const f4*)(mrow + 4);
      f4 m2 = *(const f4*)(mrow + 16);
      f4 m3 = *(const f4*)(mrow + 20);
      #pragma unroll
      for (int r = 0; r < 4; ++r) {
        s[r]      = c0[r]     + m0[r];
        s[4 + r]  = c0[4 + r] + m1[r];
        s[8 + r]  = c1[r]     + m2[r];
        s[12 + r] = c1[4 + r] + m3[r];
      }
    }

    // ---- online softmax; lanes l and l+16 share one q row -> xor-16 pair ----
    float mx = s[0];
    #pragma unroll
    for (int i = 1; i < 16; ++i) mx = fmaxf(mx, s[i]);
    mx = fmaxf(mx, __shfl_xor(mx, 16, 32));
    const float m_new = fmaxf(run_m, mx);
    const float scale = __expf(run_m - m_new);

    float p[16], ls = 0.0f;
    #pragma unroll
    for (int i = 0; i < 16; ++i) { p[i] = __expf(s[i] - m_new); ls += p[i]; }
    ls += __shfl_xor(ls, 16, 32);
    run_l = run_l * scale + ls;
    run_m = m_new;

    #pragma unroll
    for (int t = 0; t < 4; ++t)
      #pragma unroll
      for (int r = 0; r < 8; ++r) oc[t][r] *= scale;

    // ---- P^T B-fragment: S^T C-layout == B-layout, pure in-register cvt ----
    VFrag pf;
    #pragma unroll
    for (int r = 0; r < 8; ++r) {
      pf.v[r]     = (_Float16)p[r];       // subtile 0 -> k {0..7 | 8..15}
      pf.v[8 + r] = (_Float16)p[8 + r];   // subtile 1 -> k {16..23 | 24..31}
    }

    // ---- O^T += V^T . P^T ----
    #pragma unroll
    for (int t = 0; t < 4; ++t) oc[t] = wmma_f16(va[t].v, pf.v, oc[t]);

    __syncthreads();
  }

  // ---- epilogue: normalize; each lane stores 8 contiguous d per tile ----
  const float inv = 1.0f / run_l;
  float* orow = Ob + (size_t)qrow * kD + hi * 8;
  #pragma unroll
  for (int t = 0; t < 4; ++t) {
    f4 lo, hh;
    #pragma unroll
    for (int r = 0; r < 4; ++r) { lo[r] = oc[t][r] * inv; hh[r] = oc[t][4 + r] * inv; }
    *(f4*)(orow + t * 16)     = lo;   // d = 16t + hi*8 + 0..3
    *(f4*)(orow + t * 16 + 4) = hh;   // d = 16t + hi*8 + 4..7
  }
}

extern "C" void kernel_launch(void* const* d_in, const int* in_sizes, int n_in,
                              void* d_out, int out_size, void* d_ws, size_t ws_size,
                              hipStream_t stream) {
  const float* q = (const float*)d_in[0];
  const float* k = (const float*)d_in[1];
  const float* v = (const float*)d_in[2];
  const float* m = (const float*)d_in[3];
  // d_in[4] sparsity_ratio, d_in[5] maintain_heads: no effect on forward o.
  float* o = (float*)d_out;

  dim3 grid(kQ / (WAVES * QT), kB * kH);   // (32, 32)
  msa_flash_fwd_kernel<<<grid, WAVES * 32, 0, stream>>>(q, k, v, m, o);
}
